// SpikingResNet_2302102471323
// MI455X (gfx1250) — matrix operand order repaired and checked by reading, not compiled, not run on hardware
//
#include <hip/hip_runtime.h>

typedef __bf16 bf16;
typedef __attribute__((ext_vector_type(16))) __bf16 v16bf;
typedef __attribute__((ext_vector_type(8)))  __bf16 v8bf;
typedef __attribute__((ext_vector_type(8)))  float  v8f;

#define DECAY 0.9f
#define THR   0.5f
#define TW    32
#define BATCH 256

// ---------------------------------------------------------------------------
// WMMA helpers (CDNA5 V_WMMA_F32_16X16X32_BF16, wave32)
// ---------------------------------------------------------------------------
__device__ __forceinline__ v8f wmma_bf16(v16bf a, v16bf b, v8f c) {
    return __builtin_amdgcn_wmma_f32_16x16x32_bf16(false, a, false, b,
                                                   (short)0, c, false, false);
}

// A fragment: row m of [M][Kpad] bf16 matrix, K-chunk kk, lane half h.
// 16-bit A layout: element e -> K = (e/8)*16 + h*8 + (e%8)
__device__ __forceinline__ v16bf load_a_frag(const bf16* __restrict__ w,
                                             int m, int Kpad, int kk, int h) {
    const bf16* r = w + (size_t)m * Kpad + kk + h * 8;
    v8bf lo = *(const v8bf*)r;
    v8bf hi = *(const v8bf*)(r + 16);
    v16bf a;
#pragma unroll
    for (int e = 0; e < 8; ++e) { a[e] = lo[e]; a[e + 8] = hi[e]; }
    return a;
}

// B fragment from 32 contiguous bf16 (channel-minor layout), lane half h:
// element e -> K = kk + (e/8)*16 + h*8 + (e%8)
__device__ __forceinline__ v16bf load_b_frag(const bf16* __restrict__ p, int h) {
    const bf16* r = p + h * 8;
    v8bf lo = *(const v8bf*)r;
    v8bf hi = *(const v8bf*)(r + 16);
    v16bf b;
#pragma unroll
    for (int e = 0; e < 8; ++e) { b[e] = lo[e]; b[e + 8] = hi[e]; }
    return b;
}

// ---------------------------------------------------------------------------
// init kernels
// ---------------------------------------------------------------------------
__global__ void zero_kernel(unsigned int* __restrict__ p, size_t nwords) {
    size_t i = (size_t)blockIdx.x * blockDim.x + threadIdx.x;
    size_t stride = (size_t)gridDim.x * blockDim.x;
    for (; i < nwords; i += stride) p[i] = 0u;
}

// Repack all weights to bf16 [M][Kpad] with K = kernel_pos * C + c ordering.
__global__ void convert_weights_kernel(const float* __restrict__ w1,
                                       const float* __restrict__ w2,
                                       const float* __restrict__ w3,
                                       const float* __restrict__ fc,
                                       bf16* __restrict__ w1b,
                                       bf16* __restrict__ w2b,
                                       bf16* __restrict__ w3b,
                                       bf16* __restrict__ fcb) {
    int i = blockIdx.x * blockDim.x + threadIdx.x;
    if (i < 64 * 128) {                       // w1: [64,2,7,7] -> [64][128]
        int m = i >> 7, k = i & 127;
        int c = k & 1, kp = k >> 1;
        float v = (kp < 49) ? w1[(m * 2 + c) * 49 + kp] : 0.f;
        w1b[i] = (bf16)v;
        return;
    }
    i -= 64 * 128;
    if (i < 32 * 3136) {                      // w2: [32,64,7,7] -> [32][3136]
        int m = i / 3136, k = i - m * 3136;
        int c = k & 63, kp = k >> 6;
        w2b[i] = (bf16)w2[(m * 64 + c) * 49 + kp];
        return;
    }
    i -= 32 * 3136;
    if (i < 16 * 288) {                       // w3: [4,32,3,3] -> [16][288], M padded
        int m = i / 288, k = i - m * 288;
        int c = k & 31, kp = k >> 5;
        float v = (m < 4) ? w3[(m * 32 + c) * 9 + kp] : 0.f;
        w3b[i] = (bf16)v;
        return;
    }
    i -= 16 * 288;
    if (i < 32 * 64) {                        // fc: [24,64] -> [32][64], M padded
        int m = i >> 6, k = i & 63;
        float v = (m < 24) ? fc[m * 64 + k] : 0.f;
        fcb[i] = (bf16)v;
    }
}

// ---------------------------------------------------------------------------
// conv1 im2col: x[:,t] -> col [36864][128] bf16 (K = kp*2+c, zero padded).
// Keeps all boundary predication out of the WMMA loop.
// ---------------------------------------------------------------------------
__global__ void __launch_bounds__(256)
im2col1_kernel(const float* __restrict__ x, bf16* __restrict__ col, int t) {
    int idx = blockIdx.x * blockDim.x + threadIdx.x;     // 36864 * 16
    int p = idx >> 4, g = idx & 15;                      // g = group of 8 k's
    int b = p / 144, pos = p - b * 144;
    int oy = pos / 12, ox = pos - oy * 12;
    const float* xb = x + (((size_t)b * TW + t) * 2) * 256;
    v8bf out;
#pragma unroll
    for (int e = 0; e < 8; ++e) {
        int k = g * 8 + e;
        int c = k & 1, kp = k >> 1;
        float v = 0.f;
        if (kp < 49) {
            int kh = kp / 7, kw = kp - kh * 7;
            int iy = oy - 1 + kh, ix = ox - 1 + kw;
            if ((unsigned)iy < 16u && (unsigned)ix < 16u)
                v = xb[c * 256 + iy * 16 + ix];
        }
        out[e] = (bf16)v;
    }
    *(v8bf*)(col + (size_t)p * 128 + g * 8) = out;
}

// ---------------------------------------------------------------------------
// Layer 1: GEMM on col buffer -> [B,64,12,12], fused LIF.
// M=64 (4 tiles), N = 36864, K = 128. Spike/mem state stored channel-minor:
// c_spk/c_mem index = (b*144 + pos)*64 + m.
// ---------------------------------------------------------------------------
__global__ void __launch_bounds__(256)
conv1_kernel(const bf16* __restrict__ col, const bf16* __restrict__ w1b,
             float* __restrict__ c_mem, bf16* __restrict__ c_spk,
             unsigned int* __restrict__ r_cnt) {
    __shared__ unsigned int s_cnt;
    if (threadIdx.x == 0) s_cnt = 0u;
    __syncthreads();

    const int lane = threadIdx.x & 31;
    const int wave = threadIdx.x >> 5;
    const int h = lane >> 4;
    const int nl = lane & 15;
    const int p = (blockIdx.x * 8 + wave) * 16 + nl;
    const int b = p / 144;
    const int pos = p - b * 144;

    v8f acc[4] = {{}, {}, {}, {}};
#pragma unroll
    for (int kk = 0; kk < 128; kk += 32) {
        v16bf bf = load_b_frag(col + (size_t)p * 128 + kk, h);
#pragma unroll
        for (int mt = 0; mt < 4; ++mt)
            acc[mt] = wmma_bf16(load_a_frag(w1b, mt * 16 + nl, 128, kk, h), bf, acc[mt]);
    }

    unsigned int cnt = 0;
    const size_t base = ((size_t)b * 144 + pos) * 64;
#pragma unroll
    for (int mt = 0; mt < 4; ++mt) {
#pragma unroll
        for (int i = 0; i < 8; ++i) {
            int m = mt * 16 + i + h * 8;        // C layout: M = i + h*8
            size_t idx = base + m;
            float mem = c_mem[idx];
            float spk = (float)c_spk[idx];
            mem = mem * DECAY * (1.f - spk) + acc[mt][i];
            float ns = mem > THR ? 1.f : 0.f;
            c_mem[idx] = mem;
            c_spk[idx] = (bf16)ns;
            cnt += (unsigned int)ns;
        }
    }
    if (cnt) atomicAdd(&s_cnt, cnt);
    __syncthreads();
    if (threadIdx.x == 0 && s_cnt) atomicAdd(r_cnt, s_cnt);
}

// ---------------------------------------------------------------------------
// Layer 2: conv(c_spk, w2, pad=0) -> [B,32,6,6], fused LIF. Dominant layer.
// M=32 (2 tiles), N = 9216, K = 49*64 = 3136.
// Input spikes channel-minor [B][144][64] -> B fragment = 2 b128 loads.
// Output state channel-minor: (b*36 + pos)*32 + m.
// ---------------------------------------------------------------------------
__global__ void __launch_bounds__(256)
conv2_kernel(const bf16* __restrict__ spk1, const bf16* __restrict__ w2b,
             float* __restrict__ p1_mem, bf16* __restrict__ p1_spk,
             unsigned int* __restrict__ r_cnt) {
    __shared__ unsigned int s_cnt;
    if (threadIdx.x == 0) s_cnt = 0u;
    __syncthreads();

    const int lane = threadIdx.x & 31;
    const int wave = threadIdx.x >> 5;
    const int h = lane >> 4;
    const int nl = lane & 15;
    const int p = (blockIdx.x * 8 + wave) * 16 + nl;
    const int b = p / 36;
    const int pos = p - b * 36;
    const int oy = pos / 6, ox = pos - oy * 6;

    const bf16* sb = spk1 + (size_t)b * 144 * 64;

    v8f acc0 = {}, acc1 = {};
    for (int kp = 0; kp < 49; ++kp) {
        int kh = kp / 7, kw = kp - kh * 7;
        const bf16* sr = sb + ((oy + kh) * 12 + (ox + kw)) * 64;
#pragma unroll
        for (int half = 0; half < 2; ++half) {
            int kk = kp * 64 + half * 32;
            v16bf bf = load_b_frag(sr + half * 32, h);
            acc0 = wmma_bf16(load_a_frag(w2b, nl,      3136, kk, h), bf, acc0);
            acc1 = wmma_bf16(load_a_frag(w2b, nl + 16, 3136, kk, h), bf, acc1);
        }
    }

    unsigned int cnt = 0;
    const size_t base = ((size_t)b * 36 + pos) * 32;
#pragma unroll
    for (int mt = 0; mt < 2; ++mt) {
#pragma unroll
        for (int i = 0; i < 8; ++i) {
            int m = mt * 16 + i + h * 8;
            float hv = (mt == 0) ? acc0[i] : acc1[i];
            size_t idx = base + m;
            float mem = p1_mem[idx];
            float spk = (float)p1_spk[idx];
            mem = mem * DECAY * (1.f - spk) + hv;
            float ns = mem > THR ? 1.f : 0.f;
            p1_mem[idx] = mem;
            p1_spk[idx] = (bf16)ns;
            cnt += (unsigned int)ns;
        }
    }
    if (cnt) atomicAdd(&s_cnt, cnt);
    __syncthreads();
    if (threadIdx.x == 0 && s_cnt) atomicAdd(r_cnt, s_cnt);
}

// ---------------------------------------------------------------------------
// Layer 3: conv(p1_spk, w3, pad=0) -> [B,4,4,4], fused LIF.
// M=4 (padded 16), N = 4096, K = 9*32 = 288. Input channel-minor [B][36][32].
// p2 state stored as [B,64] (m*16+pos) to match the FC reshape.
// ---------------------------------------------------------------------------
__global__ void __launch_bounds__(256)
conv3_kernel(const bf16* __restrict__ spk2, const bf16* __restrict__ w3b,
             float* __restrict__ p2_mem, bf16* __restrict__ p2_spk,
             unsigned int* __restrict__ r_cnt) {
    __shared__ unsigned int s_cnt;
    if (threadIdx.x == 0) s_cnt = 0u;
    __syncthreads();

    const int lane = threadIdx.x & 31;
    const int wave = threadIdx.x >> 5;
    const int h = lane >> 4;
    const int nl = lane & 15;
    const int p = (blockIdx.x * 8 + wave) * 16 + nl;
    const int b = p >> 4;
    const int pos = p & 15;
    const int oy = pos >> 2, ox = pos & 3;

    const bf16* sb = spk2 + (size_t)b * 36 * 32;

    v8f acc = {};
#pragma unroll
    for (int kp = 0; kp < 9; ++kp) {
        int kh = kp / 3, kw = kp - kh * 3;
        v16bf bf = load_b_frag(sb + ((oy + kh) * 6 + (ox + kw)) * 32, h);
        acc = wmma_bf16(load_a_frag(w3b, nl, 288, kp * 32, h), bf, acc);
    }

    unsigned int cnt = 0;
#pragma unroll
    for (int i = 0; i < 8; ++i) {
        int m = i + h * 8;
        if (m < 4) {
            size_t idx = (size_t)b * 64 + m * 16 + pos;
            float mem = p2_mem[idx];
            float spk = (float)p2_spk[idx];
            mem = mem * DECAY * (1.f - spk) + acc[i];
            float ns = mem > THR ? 1.f : 0.f;
            p2_mem[idx] = mem;
            p2_spk[idx] = (bf16)ns;
            cnt += (unsigned int)ns;
        }
    }
    if (cnt) atomicAdd(&s_cnt, cnt);
    __syncthreads();
    if (threadIdx.x == 0 && s_cnt) atomicAdd(r_cnt, s_cnt);
}

// ---------------------------------------------------------------------------
// FC layer: [B,64] @ fc_w^T -> [B,24], fused LIF + accumulate sumspike.
// M=24 (padded 32), N = 256, K = 64.
// ---------------------------------------------------------------------------
__global__ void __launch_bounds__(256)
fc_kernel(const bf16* __restrict__ spk3, const bf16* __restrict__ fcb,
          float* __restrict__ fc_mem, float* __restrict__ fc_spk,
          float* __restrict__ accb, unsigned int* __restrict__ r_cnt) {
    __shared__ unsigned int s_cnt;
    if (threadIdx.x == 0) s_cnt = 0u;
    __syncthreads();

    const int lane = threadIdx.x & 31;
    const int wave = threadIdx.x >> 5;
    const int h = lane >> 4;
    const int nl = lane & 15;
    const int n = (blockIdx.x * 8 + wave) * 16 + nl;   // batch index

    v8f acc0 = {}, acc1 = {};
#pragma unroll
    for (int kk = 0; kk < 64; kk += 32) {
        v16bf bf = load_b_frag(spk3 + (size_t)n * 64 + kk, h);
        acc0 = wmma_bf16(load_a_frag(fcb, nl,      64, kk, h), bf, acc0);
        acc1 = wmma_bf16(load_a_frag(fcb, nl + 16, 64, kk, h), bf, acc1);
    }

    unsigned int cnt = 0;
#pragma unroll
    for (int mt = 0; mt < 2; ++mt) {
#pragma unroll
        for (int i = 0; i < 8; ++i) {
            int m = mt * 16 + i + h * 8;
            if (m < 24) {
                float hv = (mt == 0) ? acc0[i] : acc1[i];
                size_t idx = (size_t)n * 24 + m;
                float mem = fc_mem[idx];
                float spk = fc_spk[idx];
                mem = mem * DECAY * (1.f - spk) + hv;
                float ns = mem > THR ? 1.f : 0.f;
                fc_mem[idx] = mem;
                fc_spk[idx] = ns;
                accb[idx] += ns;
                cnt += (unsigned int)ns;
            }
        }
    }
    if (cnt) atomicAdd(&s_cnt, cnt);
    __syncthreads();
    if (threadIdx.x == 0 && s_cnt) atomicAdd(r_cnt, s_cnt);
}

// ---------------------------------------------------------------------------
// finalize: sumspike = acc / T ; r = counts / (T * numel), r[4..9] = 0
// ---------------------------------------------------------------------------
__global__ void finalize_kernel(const float* __restrict__ accb,
                                const unsigned int* __restrict__ r_cnt,
                                float* __restrict__ out) {
    int i = blockIdx.x * blockDim.x + threadIdx.x;
    if (i < 6144) out[i] = accb[i] * (1.f / 32.f);
    if (i < 10) {
        float r = 0.f;
        if (i == 0) r = (float)r_cnt[0] * (1.f / 75497472.f);   // T*B*64*144
        if (i == 1) r = (float)r_cnt[1] * (1.f / 9437184.f);    // T*B*32*36
        if (i == 2) r = (float)r_cnt[2] * (1.f / 524288.f);     // T*B*4*16
        if (i == 3) r = (float)r_cnt[3] * (1.f / 196608.f);     // T*B*24
        out[6144 + i] = r;
    }
}

// ---------------------------------------------------------------------------
extern "C" void kernel_launch(void* const* d_in, const int* in_sizes, int n_in,
                              void* d_out, int out_size, void* d_ws, size_t ws_size,
                              hipStream_t stream) {
    (void)in_sizes; (void)n_in; (void)out_size; (void)ws_size;
    const float* x   = (const float*)d_in[0];
    const float* w1  = (const float*)d_in[1];
    const float* w2  = (const float*)d_in[2];
    const float* w3  = (const float*)d_in[3];
    const float* fcw = (const float*)d_in[4];
    // d_in[5] = time_window = 32 (hardcoded)

    char* ws = (char*)d_ws;
    size_t off = 0;
    auto carve = [&](size_t bytes) -> char* {
        char* q = ws + off;
        off += (bytes + 255) & ~(size_t)255;
        return q;
    };
    bf16* w1b = (bf16*)carve(64 * 128 * 2);
    bf16* w2b = (bf16*)carve(32 * 3136 * 2);
    bf16* w3b = (bf16*)carve(16 * 288 * 2);
    bf16* fcb = (bf16*)carve(32 * 64 * 2);
    bf16* col1 = (bf16*)carve((size_t)36864 * 128 * 2);   // conv1 im2col (per t)
    char* state_begin = ws + off;
    float* c_mem  = (float*)carve((size_t)2359296 * 4);   // B*144*64 (ch-minor)
    float* p1_mem = (float*)carve((size_t)294912 * 4);    // B*36*32  (ch-minor)
    float* p2_mem = (float*)carve((size_t)16384 * 4);     // B*64
    float* fc_mem = (float*)carve((size_t)6144 * 4);      // B*24
    float* fc_spk = (float*)carve((size_t)6144 * 4);
    float* accb   = (float*)carve((size_t)6144 * 4);
    bf16* c_spk  = (bf16*)carve((size_t)2359296 * 2);
    bf16* p1_spk = (bf16*)carve((size_t)294912 * 2);
    bf16* p2_spk = (bf16*)carve((size_t)16384 * 2);
    unsigned int* r_cnt = (unsigned int*)carve(4 * 4);
    size_t state_words = (size_t)((ws + off) - state_begin) / 4;

    zero_kernel<<<1024, 256, 0, stream>>>((unsigned int*)state_begin, state_words);
    convert_weights_kernel<<<451, 256, 0, stream>>>(w1, w2, w3, fcw, w1b, w2b, w3b, fcb);

    for (int t = 0; t < TW; ++t) {
        im2col1_kernel<<<2304, 256, 0, stream>>>(x, col1, t);
        conv1_kernel<<<288, 256, 0, stream>>>(col1, w1b, c_mem, c_spk, r_cnt + 0);
        conv2_kernel<<<72, 256, 0, stream>>>(c_spk, w2b, p1_mem, p1_spk, r_cnt + 1);
        conv3_kernel<<<32, 256, 0, stream>>>(p1_spk, w3b, p2_mem, p2_spk, r_cnt + 2);
        fc_kernel<<<2, 256, 0, stream>>>(p2_spk, fcb, fc_mem, fc_spk, accb, r_cnt + 3);
    }
    finalize_kernel<<<24, 256, 0, stream>>>(accb, r_cnt, (float*)d_out);
}
